// BiLSTM_65111704207396
// MI455X (gfx1250) — compile-verified, matrix operand order checked by reading, not run on hardware
//
#include <hip/hip_runtime.h>
#include <hip/hip_bf16.h>

// ---------------------------------------------------------------------------
// Bidirectional LSTM for MI455X (gfx1250, wave32, WMMA 16x16x32 bf16).
//
// Stage 0: pack W1/W2 (fp32 [768,1024]) into bf16 WMMA B-fragments.
// Stage 1: input GEMM G[dir][m=t*32+b][n] = x @ Wx + bias (time-hoisted, bwd
//          stored time-reversed). B-fragments staged into LDS per block via
//          async-to-LDS double buffering (8 waves share each fragment).
// Stage 2: persistent scan, 1 WG of 1024 threads per direction. h lives in
//          LDS as ready A-fragments, c in VGPRs, Wh streams from L2, next
//          timestep's G prefetched into registers to hide HBM latency on the
//          serial recurrence path.
// ---------------------------------------------------------------------------

typedef __attribute__((ext_vector_type(16))) __bf16 v16bf;
typedef __attribute__((ext_vector_type(8)))  float  v8f;

// exact pointee type the async-to-LDS builtin wants (from clang diagnostic):
typedef int v4i_ __attribute__((vector_size(16)));
typedef __attribute__((address_space(1))) v4i_* g_v4i_p;   // global (AS1)
typedef __attribute__((address_space(3))) v4i_* l_v4i_p;   // LDS (AS3)

#define T_LEN 1024
#define BATCH 32
#define IN    512
#define HID   256
#define NG    1024                      // 4*HID
#define WX_ELEMS_PER_DIR  (64*16*32*16) // 524288 bf16
#define WH_ELEMS_PER_DIR  (64*8*32*16)  // 262144 bf16
#define WX_FRAGS_PER_DIR  (64*16*32)
#define WH_FRAGS_PER_DIR  (64*8*32)
#define G_ELEMS_PER_DIR   ((size_t)32768 * (size_t)NG)

// -------- async-to-LDS availability probe (compile-safe either way) --------
#ifndef __has_builtin
#define __has_builtin(x) 0
#endif
#if __has_builtin(__builtin_amdgcn_global_load_async_to_lds_b128)
#define HAVE_ASYNC_LDS 1
#else
#define HAVE_ASYNC_LDS 0
#endif

#if HAVE_ASYNC_LDS
#if __has_builtin(__builtin_amdgcn_s_wait_asynccnt)
#define WAIT_ASYNC(n) __builtin_amdgcn_s_wait_asynccnt(n)
#else
#define WAIT_ASYNC(n) asm volatile("s_wait_asynccnt %0" :: "i"(n) : "memory")
#endif
#else
#define WAIT_ASYNC(n) ((void)0)
#endif

__device__ __forceinline__ float sigmoidf_(float x) {
    return 1.0f / (1.0f + __expf(-x));
}

// Copy BYTES (global->LDS), 16B per thread per sweep, 4 KB stride.
// Async path: ASYNCcnt-tracked direct-to-LDS DMA, no VGPR staging.
template <int BYTES>
__device__ __forceinline__ void stage_group(const char* gsrc, char* ldst, int tid) {
    const char* s = gsrc + tid * 16;
    char* d = ldst + tid * 16;
#if HAVE_ASYNC_LDS
    #pragma unroll
    for (int i = 0; i < BYTES / 4096; ++i) {
        __builtin_amdgcn_global_load_async_to_lds_b128(
            (g_v4i_p)(uintptr_t)(s + i * 4096),
            (l_v4i_p)(uint32_t)(uintptr_t)(d + i * 4096),
            0, 0);
    }
#else
    #pragma unroll
    for (int i = 0; i < BYTES / 4096; ++i)
        *(uint4*)(d + i * 4096) = *(const uint4*)(s + i * 4096);
#endif
}

// --------------------------------------------------------------------- pack
__global__ void __launch_bounds__(256)
pack_w_kernel(const float* __restrict__ W1, const float* __restrict__ W2,
              __bf16* __restrict__ wxp, __bf16* __restrict__ whp) {
    const int perdir = WX_ELEMS_PER_DIR + WH_ELEMS_PER_DIR; // 786432
    int tid = blockIdx.x * blockDim.x + threadIdx.x;
    if (tid >= 2 * perdir) return;
    int d = tid / perdir;
    int r = tid - d * perdir;
    const float* W = d ? W2 : W1;
    if (r < WX_ELEMS_PER_DIR) {
        int e    = r & 15;
        int lane = (r >> 4) & 31;
        int kt   = (r >> 9) & 15;
        int nt   = r >> 13;
        int k = kt * 32 + ((lane >> 4) << 4) + e;   // 0..511
        int n = nt * 16 + (lane & 15);              // 0..1023
        wxp[(size_t)d * WX_ELEMS_PER_DIR + r] = (__bf16)W[k * NG + n];
    } else {
        int idx  = r - WX_ELEMS_PER_DIR;
        int e    = idx & 15;
        int lane = (idx >> 4) & 31;
        int kt   = (idx >> 9) & 7;
        int nt   = idx >> 12;
        int k = kt * 32 + ((lane >> 4) << 4) + e;   // 0..255
        int n = nt * 16 + (lane & 15);
        whp[(size_t)d * WH_ELEMS_PER_DIR + idx] = (__bf16)W[(IN + k) * NG + n];
    }
}

// ---------------------------------------------------------- input projection
// Block = 256 threads = 8 waves, each wave owns one 16-row M-tile (A frags in
// 128 VGPRs, reused across all 64 N-tiles). All 8 waves consume the SAME
// B-fragments, so they are staged through LDS: 2 N-tiles per group (32 KB),
// double-buffered, async copy of group g+1 overlapping WMMAs of group g.
#define NT_GRP    2
#define GRP_BYTES (NT_GRP * 16 * 32 * 32)   // 32768
#define N_GRPS    (64 / NT_GRP)             // 32

__global__ void __launch_bounds__(256)
input_gemm_kernel(const float* __restrict__ x,
                  const float* __restrict__ b1, const float* __restrict__ b2,
                  const __bf16* __restrict__ wxp, float* __restrict__ G) {
    __shared__ __align__(128) char bstage[2][GRP_BYTES];   // 64 KB

    int tid  = threadIdx.x;
    int lane = tid & 31;
    int wv   = tid >> 5;                       // 0..7
    int blk  = blockIdx.x;                     // 0..511
    int dir  = blk >> 8;                       // 256 blocks per dir
    int mt   = (blk & 255) * 8 + wv;           // this wave's M-tile

    int m    = mt * 16 + (lane & 15);          // G row (t*32 + b)
    int tg   = m >> 5;
    int b    = m & 31;
    int tsrc = dir ? (T_LEN - 1 - tg) : tg;    // bwd reads reversed x
    const float* xrow = x + ((size_t)b * T_LEN + tsrc) * IN;
    int khalf = (lane >> 4) << 4;              // lanes 16-31: +16 in K

    v16bf afrag[16];
    #pragma unroll
    for (int kt = 0; kt < 16; ++kt) {
        const float4* p4 = (const float4*)(xrow + kt * 32 + khalf);
        v16bf a;
        #pragma unroll
        for (int q = 0; q < 4; ++q) {
            float4 v = p4[q];
            a[4 * q + 0] = (__bf16)v.x;
            a[4 * q + 1] = (__bf16)v.y;
            a[4 * q + 2] = (__bf16)v.z;
            a[4 * q + 3] = (__bf16)v.w;
        }
        afrag[kt] = a;
    }

    const float* bias = dir ? b2 : b1;
    const char*  bsrc = (const char*)(wxp + (size_t)dir * WX_ELEMS_PER_DIR);
    float* Gp = G + (size_t)dir * G_ELEMS_PER_DIR;
    int nlo   = lane & 15;
    int mbase = mt * 16 + ((lane >> 4) << 3);  // C-layout row base

    stage_group<GRP_BYTES>(bsrc, bstage[0], tid);          // prologue
    for (int grp = 0; grp < N_GRPS; ++grp) {
        int cur = grp & 1;
        if (grp + 1 < N_GRPS) {
            stage_group<GRP_BYTES>(bsrc + (size_t)(grp + 1) * GRP_BYTES,
                                   bstage[cur ^ 1], tid);
            WAIT_ASYNC(GRP_BYTES / 4096);      // own copies of grp are done
        } else {
            WAIT_ASYNC(0);
        }
        __syncthreads();                       // everyone's grp copies landed

        const v16bf* bs = (const v16bf*)&bstage[cur][0];
        #pragma unroll
        for (int ntl = 0; ntl < NT_GRP; ++ntl) {
            int nt = grp * NT_GRP + ntl;
            float bv = bias[nt * 16 + nlo];
            v8f acc;
            #pragma unroll
            for (int r = 0; r < 8; ++r) acc[r] = bv;   // fold bias into G
            #pragma unroll
            for (int kt = 0; kt < 16; ++kt) {
                v16bf bf = bs[(ntl * 16 + kt) * 32 + lane];
                acc = __builtin_amdgcn_wmma_f32_16x16x32_bf16(
                    false, afrag[kt], false, bf, (short)0, acc, false, false);
            }
            int n = nt * 16 + nlo;
            #pragma unroll
            for (int r = 0; r < 8; ++r)
                Gp[(size_t)(mbase + r) * NG + n] = acc[r];
        }
        __syncthreads();                       // done reading before overwrite
    }
}

// ------------------------------------------------------------------ the scan
// grid = 2 (one WG per direction), block = 1024 (32 waves on one WGP).
// Wave w: mt = w&1 (batch half), nh = w>>1 (16 hidden units). Its 4 N-tiles
// are the i/g/f/o gate columns for those hidden units, so each lane ends up
// holding all four gate pre-activations for its (b, j) in registers.
// Next timestep's G rows are prefetched into registers during the current
// step so HBM latency stays off the serial recurrence path.
__global__ void __launch_bounds__(1024)
lstm_scan_kernel(const float* __restrict__ G, const __bf16* __restrict__ whp,
                 float* __restrict__ out) {
    int dir  = blockIdx.x;
    int lane = threadIdx.x & 31;
    int w    = threadIdx.x >> 5;
    int mt   = w & 1;
    int nh   = w >> 1;                                 // 0..15

    // h state stored directly as WMMA A-fragments: [mt][kt][lane][e], bf16
    __shared__ __align__(32) __bf16 hfrag[2][8][32][16];   // 16 KB
    for (int i = threadIdx.x; i < 2 * 8 * 32 * 16; i += 1024)
        ((__bf16*)hfrag)[i] = (__bf16)0.0f;            // h0 = 0
    __syncthreads();

    const float* Gp  = G + (size_t)dir * G_ELEMS_PER_DIR;
    const v16bf* whf = ((const v16bf*)whp) + (size_t)dir * WH_FRAGS_PER_DIR;

    float creg[8];                                     // c state: pure VGPR
    #pragma unroll
    for (int r = 0; r < 8; ++r) creg[r] = 0.0f;

    int nlo   = lane & 15;
    int mbase = mt * 16 + ((lane >> 4) << 3);          // + r -> batch b
    int j     = nh * 16 + nlo;                         // this lane's hidden j
    int ktj   = j >> 5;                                // h-fragment K-tile
    int lhp   = (j & 16) ? 16 : 0;                     // fragment lane-half
    int ej    = j & 15;

    // prefetch G rows for t=0
    v8f gnext[4];
    #pragma unroll
    for (int g = 0; g < 4; ++g) {
        int n = (g * 16 + nh) * 16 + nlo;
        const float* grow = Gp + (size_t)mbase * NG + n;
        #pragma unroll
        for (int r = 0; r < 8; ++r) gnext[g][r] = grow[(size_t)r * NG];
    }

    for (int t = 0; t < T_LEN; ++t) {
        // consume prefetched x@Wx + bias
        v8f acc[4];
        #pragma unroll
        for (int g = 0; g < 4; ++g) acc[g] = gnext[g];

        // issue next step's G loads now; consumed only next iteration, so
        // their HBM latency hides behind the WMMAs + barriers + gate math
        int tn = (t + 1 < T_LEN) ? (t + 1) : t;
        #pragma unroll
        for (int g = 0; g < 4; ++g) {
            int n = (g * 16 + nh) * 16 + nlo;
            const float* grow = Gp + ((size_t)tn * 32 + mbase) * NG + n;
            #pragma unroll
            for (int r = 0; r < 8; ++r) gnext[g][r] = grow[(size_t)r * NG];
        }

        // recurrent GEMM: kt outer so one A-fragment live, 4 indep. WMMAs
        #pragma unroll
        for (int kt = 0; kt < 8; ++kt) {
            v16bf af = *(const v16bf*)&hfrag[mt][kt][lane][0];
            #pragma unroll
            for (int g = 0; g < 4; ++g) {
                v16bf bf = whf[((g * 16 + nh) * 8 + kt) * 32 + lane];
                acc[g] = __builtin_amdgcn_wmma_f32_16x16x32_bf16(
                    false, af, false, bf, (short)0, acc[g], false, false);
            }
        }
        __syncthreads();   // everyone consumed hfrag for this step

        // gates fully in registers: acc[0]=i acc[1]=g acc[2]=f acc[3]=o
        int tout = dir ? (T_LEN - 1 - t) : t;
        #pragma unroll
        for (int r = 0; r < 8; ++r) {
            float iv = acc[0][r], gv = acc[1][r], fv = acc[2][r], ov = acc[3][r];
            float c = creg[r];
            float f = sigmoidf_(fv + 1.0f);            // haiku forget bias +1
            c = f * c + sigmoidf_(iv) * tanhf(gv);
            float h = sigmoidf_(ov) * tanhf(c);
            creg[r] = c;
            int b = mbase + r;
            hfrag[mt][ktj][(b & 15) + lhp][ej] = (__bf16)h;  // next A-frag
            out[((size_t)b * T_LEN + tout) * (2 * HID) + dir * HID + j] = h;
        }
        __syncthreads();   // publish h(t) for step t+1
    }
}

// ------------------------------------------------------------------- launch
extern "C" void kernel_launch(void* const* d_in, const int* in_sizes, int n_in,
                              void* d_out, int out_size, void* d_ws, size_t ws_size,
                              hipStream_t stream) {
    const float* x  = (const float*)d_in[0];
    const float* W1 = (const float*)d_in[1];
    const float* b1 = (const float*)d_in[2];
    const float* W2 = (const float*)d_in[3];
    const float* b2 = (const float*)d_in[4];
    float* out = (float*)d_out;

    // workspace layout (all 256B-aligned): G fp32, then bf16 weight packs
    float*  G   = (float*)d_ws;
    __bf16* wxp = (__bf16*)((char*)d_ws + 2 * G_ELEMS_PER_DIR * sizeof(float));
    __bf16* whp = (__bf16*)((char*)wxp + 2 * WX_ELEMS_PER_DIR * sizeof(__bf16));

    pack_w_kernel<<<6144, 256, 0, stream>>>(W1, W2, wxp, whp);
    input_gemm_kernel<<<512, 256, 0, stream>>>(x, b1, b2, wxp, G);
    lstm_scan_kernel<<<2, 1024, 0, stream>>>(G, whp, out);
}
